// SparseSINConv_4372276707362
// MI455X (gfx1250) — compile-verified, matrix operand order checked by reading, not run on hardware
//
#include <hip/hip_runtime.h>
#include <hip/hip_bf16.h>

typedef __attribute__((ext_vector_type(16))) __bf16 v16bf;
typedef __attribute__((ext_vector_type(8)))  __bf16 v8bf;
typedef __attribute__((ext_vector_type(8)))  float  v8f;

#define D 128
#define TILE_ROWS 128
#define KCHUNK 32
#define STF 40   // f32 LDS row stride: 32 + 8 pad (160B, keeps 16B alignment)

static __device__ __forceinline__ unsigned bfround(float f) {
    unsigned u = __builtin_bit_cast(unsigned, f);
    return u + 0x7FFFu + ((u >> 16) & 1u);    // round-to-nearest-even (take bits [31:16])
}

static __device__ __forceinline__ __bf16 f2bf(float f) {
    unsigned short s = (unsigned short)(bfround(f) >> 16);
    return __builtin_bit_cast(__bf16, s);
}

// pack two f32 -> dword of 2x bf16 (low = a, high = b)
static __device__ __forceinline__ unsigned pk_bf16(float a, float b) {
#if __has_builtin(__builtin_amdgcn_cvt_pk_bf16_f32)
    return __builtin_bit_cast(unsigned, __builtin_amdgcn_cvt_pk_bf16_f32(a, b));
#else
    return (bfround(a) >> 16) | (bfround(b) & 0xFFFF0000u);
#endif
}

// ---------------- elementwise helpers ----------------

__global__ void zero256_kernel(float* p) { p[threadIdx.x] = 0.f; }

__global__ void copy_f4_kernel(const float4* __restrict__ src, float4* __restrict__ dst, long n4) {
    long i = (long)blockIdx.x * blockDim.x + threadIdx.x;
    if (i < n4) dst[i] = src[i];
}

// out[tgt[e]] += x[src[e]]  (8 threads per edge, 16 feats each)
__global__ void scatter_add_kernel(const float* __restrict__ x, const int* __restrict__ src,
                                   const int* __restrict__ tgt, float* __restrict__ out, long E) {
    long i = (long)blockIdx.x * blockDim.x + threadIdx.x;
    long e = i >> 3;
    if (e >= E) return;
    int c = (int)(i & 7) * 16;
    long s = src[e], t = tgt[e];
    const float4* xs = (const float4*)(x + s * D + c);
    float* o = out + t * D + c;
#pragma unroll
    for (int q = 0; q < 4; ++q) {
        float4 v = xs[q];
        atomicAdd(o + 4 * q + 0, v.x);
        atomicAdd(o + 4 * q + 1, v.y);
        atomicAdd(o + 4 * q + 2, v.z);
        atomicAdd(o + 4 * q + 3, v.w);
    }
}

// scale = g * rsqrt(var+eps); shift = beta - mu*scale (slot: [0..255]=sum/sumsq)
__global__ void finalize_bn_kernel(float* __restrict__ slot, const float* __restrict__ g,
                                   const float* __restrict__ be, float invN) {
    int c = threadIdx.x;  // 128
    float mu  = slot[c] * invN;
    float var = slot[128 + c] * invN - mu * mu;
    float sc  = g[c] * rsqrtf(var + 1e-5f);
    slot[256 + c] = sc;
    slot[384 + c] = be[c] - mu * sc;
}

// y = relu(h*scale + shift), vectorized float4
__global__ void affine_relu_kernel(const float* __restrict__ h, const float* __restrict__ slot,
                                   float* __restrict__ y, long n4) {
    long i = (long)blockIdx.x * blockDim.x + threadIdx.x;
    if (i >= n4) return;
    const float* scale = slot + 256;
    const float* shift = slot + 384;
    int c = (int)(i & 31) << 2;
    float4 v = ((const float4*)h)[i];
    v.x = fmaxf(v.x * scale[c + 0] + shift[c + 0], 0.f);
    v.y = fmaxf(v.y * scale[c + 1] + shift[c + 1], 0.f);
    v.z = fmaxf(v.z * scale[c + 2] + shift[c + 2], 0.f);
    v.w = fmaxf(v.w * scale[c + 3] + shift[c + 3], 0.f);
    ((float4*)y)[i] = v;
}

// w[k*128 + n] (f32) -> wt[n*K + k] (bf16)
__global__ void wconv_kernel(const float* __restrict__ w, __bf16* __restrict__ wt, int K) {
    int i = blockIdx.x * blockDim.x + threadIdx.x;
    if (i >= K * D) return;
    int k = i / D, n = i % D;
    wt[(size_t)n * K + k] = f2bf(w[i]);
}

// ---------------- fused GEMM + BN-stats ----------------
// out[N,128] = relu_affine(in)[N,Ktot] @ W[Ktot,128] + bias ; column sum/sumsq into gstats[0..255].
// A chunks staged f32 via async global->LDS DMA; BN-affine+ReLU+bf16 round fused into fragment build.
// Ktot=256 reads k<128 from in0 (slot0 affine) and k>=128 from in1 (slot1 affine).
__global__ __launch_bounds__(256)
void gemm_bn_kernel(const float* __restrict__ in0, const float* __restrict__ slot0,
                    const float* __restrict__ in1, const float* __restrict__ slot1,
                    const __bf16* __restrict__ wt, const float* __restrict__ bias,
                    float* __restrict__ out, float* __restrict__ gstats,
                    int N, int Ktot)
{
    __shared__ float lAf[TILE_ROWS * STF];   // f32 A chunk, async-DMA destination
    __shared__ float lAff[2 * 256];          // per-seg [scale 128 | shift 128]
    __shared__ float lstats[256];

    const int tid     = threadIdx.x;
    const int wave    = tid >> 5;
    const int lane    = tid & 31;
    const int halfSel = lane >> 4;
    const int l16     = lane & 15;
    const int base    = halfSel * 8;
    const int rowBlock = blockIdx.x * TILE_ROWS;

    lstats[tid] = 0.f;
    if (tid < 128) {   // preload BN affine params (identity if no producer BN)
        lAff[tid]       = slot0 ? slot0[256 + tid] : 1.f;
        lAff[128 + tid] = slot0 ? slot0[384 + tid] : 0.f;
        lAff[256 + tid] = slot1 ? slot1[256 + tid] : 1.f;
        lAff[384 + tid] = slot1 ? slot1[384 + tid] : 0.f;
    }

    v8f acc[8];
#pragma unroll
    for (int j = 0; j < 8; ++j) acc[j] = (v8f)0.0f;

    const bool have0 = (slot0 != nullptr);
    const bool have1 = (slot1 != nullptr);

    for (int k0 = 0; k0 < Ktot; k0 += KCHUNK) {
        const float* src;
        int seg, klocal;
        if (k0 < 128) { src = in0; seg = 0; klocal = k0; }
        else          { src = in1; seg = 1; klocal = k0 - 128; }
        const bool doAff = seg ? have1 : have0;

        __syncthreads();   // prior chunk's LDS reads complete (consumed by WMMA) before overwrite
        {   // async stage: thread -> (row, 16-float half), 4 x 16B DMA transfers
            int r    = tid >> 1;
            int h16  = (tid & 1) * 16;
            int grow = rowBlock + r;
            float* ldst = &lAf[r * STF + h16];
            if (grow < N) {
                unsigned lo = (unsigned)(uintptr_t)ldst;   // low 32 bits = LDS offset
                unsigned long long ga =
                    (unsigned long long)(uintptr_t)(src + (size_t)grow * D + klocal + h16);
                asm volatile(
                    "global_load_async_to_lds_b128 %0, %1, off\n\t"
                    "global_load_async_to_lds_b128 %0, %1, off offset:16\n\t"
                    "global_load_async_to_lds_b128 %0, %1, off offset:32\n\t"
                    "global_load_async_to_lds_b128 %0, %1, off offset:48"
                    :: "v"(lo), "v"(ga) : "memory");
            } else {
#pragma unroll
                for (int q = 0; q < 16; ++q) ldst[q] = 0.f;
            }
        }
        asm volatile("s_wait_asynccnt 0x0" ::: "memory");
        __syncthreads();

        // build A fragment (ISA 16-bit layout): lane<16 K{0..7,16..23}, lane>=16 K{8..15,24..31}
        int arow = wave * 16 + l16;
        const float* ap = &lAf[arow * STF];
        float va[16];
        {
            float4 t0 = *(const float4*)(ap + base);
            float4 t1 = *(const float4*)(ap + base + 4);
            float4 t2 = *(const float4*)(ap + 16 + base);
            float4 t3 = *(const float4*)(ap + 16 + base + 4);
            va[0] = t0.x;  va[1] = t0.y;  va[2]  = t0.z;  va[3]  = t0.w;
            va[4] = t1.x;  va[5] = t1.y;  va[6]  = t1.z;  va[7]  = t1.w;
            va[8] = t2.x;  va[9] = t2.y;  va[10] = t2.z;  va[11] = t2.w;
            va[12] = t3.x; va[13] = t3.y; va[14] = t3.z;  va[15] = t3.w;
        }
        if (doAff) {
            const float* sc = &lAff[seg * 256];
            const float* sh = sc + 128;
#pragma unroll
            for (int q = 0; q < 8; ++q) {
                int c0 = klocal + base + q;
                int c1 = klocal + 16 + base + q;
                float u0 = va[q] * sc[c0] + sh[c0];
                float u1 = va[8 + q] * sc[c1] + sh[c1];
                va[q]     = u0 > 0.f ? u0 : 0.f;
                va[8 + q] = u1 > 0.f ? u1 : 0.f;
            }
        }
        union { v16bf v; unsigned u[8]; } ua;
#pragma unroll
        for (int q = 0; q < 8; ++q) ua.u[q] = pk_bf16(va[2 * q], va[2 * q + 1]);

#pragma unroll
        for (int j = 0; j < 8; ++j) {   // 8 column tiles -> full 128-wide output
            int ncol = j * 16 + l16;
            const __bf16* wp = wt + (size_t)ncol * Ktot + k0 + base;
            union { v16bf v; v8bf h[2]; } ub;
            ub.h[0] = *(const v8bf*)(wp);
            ub.h[1] = *(const v8bf*)(wp + 16);
            acc[j] = __builtin_amdgcn_wmma_f32_16x16x32_bf16(
                false, ua.v, false, ub.v, (short)0, acc[j], false, false);
        }
    }

    // epilogue: bias, store H, accumulate per-column sum/sumsq
    int rowBaseW = rowBlock + wave * 16 + halfSel * 8;
#pragma unroll
    for (int j = 0; j < 8; ++j) {
        int ncol = j * 16 + l16;
        float b = bias[ncol];
        float s = 0.f, q = 0.f;
#pragma unroll
        for (int i = 0; i < 8; ++i) {
            int grow = rowBaseW + i;
            float v = acc[j][i] + b;
            if (grow < N) {
                out[(size_t)grow * D + ncol] = v;
                s += v; q += v * v;
            }
        }
        atomicAdd(&lstats[ncol], s);
        atomicAdd(&lstats[128 + ncol], q);
    }
    __syncthreads();
    atomicAdd(&gstats[tid], lstats[tid]);
}

// ---------------- host orchestration ----------------

static inline void launch_copy(hipStream_t s, const float* src, float* dst, long n) {
    long n4 = n / 4;
    copy_f4_kernel<<<dim3((unsigned)((n4 + 255) / 256)), 256, 0, s>>>((const float4*)src, (float4*)dst, n4);
}
static inline void launch_scatter(hipStream_t s, const float* x, const int* si, const int* ti,
                                  float* out, long E) {
    long th = E * 8;
    scatter_add_kernel<<<dim3((unsigned)((th + 255) / 256)), 256, 0, s>>>(x, si, ti, out, E);
}
static inline void launch_gemm(hipStream_t s, const float* in0, const float* sl0,
                               const float* in1, const float* sl1,
                               const __bf16* wt, const float* bias,
                               float* out, float* slot, int N, int Ktot) {
    zero256_kernel<<<1, 256, 0, s>>>(slot);
    gemm_bn_kernel<<<dim3((unsigned)((N + TILE_ROWS - 1) / TILE_ROWS)), 256, 0, s>>>(
        in0, sl0, in1, sl1, wt, bias, out, slot, N, Ktot);
}

extern "C" void kernel_launch(void* const* d_in, const int* in_sizes, int n_in,
                              void* d_out, int out_size, void* d_ws, size_t ws_size,
                              hipStream_t stream) {
    const float* x0 = (const float*)d_in[0];
    const float* x1 = (const float*)d_in[1];
    const float* x2 = (const float*)d_in[2];
    const int* up_src0   = (const int*)d_in[3];
    const int* up_tgt0   = (const int*)d_in[4];
    const int* up_src1   = (const int*)d_in[5];
    const int* up_tgt1   = (const int*)d_in[6];
    const int* face_src1 = (const int*)d_in[7];
    const int* face_tgt1 = (const int*)d_in[8];
    const int* face_src2 = (const int*)d_in[9];
    const int* face_tgt2 = (const int*)d_in[10];

    const int N0 = in_sizes[0] / D, N1 = in_sizes[1] / D, N2 = in_sizes[2] / D;
    const long E0 = in_sizes[3], E1 = in_sizes[5], F1 = in_sizes[7], F2 = in_sizes[9];

    // params flattened in insertion order: u0,u1,f1,f2 (w1,b1,g1,be1,w2,b2,g2,be2), c0,c1,c2 (w,b,g,be)
    #define P(i) ((const float*)d_in[(i)])
    const int LW[11]  = {11, 15, 43, 19, 23, 27, 31, 47, 35, 39, 51};
    const int LB[11]  = {12, 16, 44, 20, 24, 28, 32, 48, 36, 40, 52};
    const int LG[11]  = {13, 17, 45, 21, 25, 29, 33, 49, 37, 41, 53};
    const int LBE[11] = {14, 18, 46, 22, 26, 30, 34, 50, 38, 42, 54};
    const int LK[11]  = {128,128,128,128,128,128,128,256,128,128,128};

    // workspace layout
    char* ws = (char*)d_ws;
    float* stats = (float*)ws;                       // 11 slots x 512 floats
    size_t off = 11 * 512 * sizeof(float);           // 22528 (256B aligned)
    __bf16* wt[11];
    {
        __bf16* wbase = (__bf16*)(ws + off);
        size_t h = 0;
        for (int i = 0; i < 11; ++i) { wt[i] = wbase + h; h += (size_t)LK[i] * D; }
        off += h * sizeof(__bf16);
        off = (off + 255) & ~(size_t)255;
    }
    const size_t bufElems = (size_t)N1 * D;
    float* bufA = (float*)(ws + off);
    float* bufB = bufA + bufElems;
    float* bufC = bufB + bufElems;
    (void)ws_size; (void)n_in; (void)out_size;

    float* y0 = (float*)d_out;
    float* y1 = y0 + (size_t)N0 * D;
    float* y2 = y1 + (size_t)N1 * D;

    // convert all weights to bf16, transposed [n][k]
    for (int i = 0; i < 11; ++i) {
        int tot = LK[i] * D;
        wconv_kernel<<<dim3((tot + 255) / 256), 256, 0, stream>>>(P(LW[i]), wt[i], LK[i]);
    }

    #define SLOT(i) (stats + (size_t)(i) * 512)
    #define FIN(i, g, be, N) finalize_bn_kernel<<<1, 128, 0, stream>>>(SLOT(i), P(g), P(be), 1.0f / (float)(N))

    // ---- dim 0: t0 = x0 + seg(x0[up_src0], up_tgt0) ----
    launch_copy(stream, x0, bufA, (long)N0 * D);
    launch_scatter(stream, x0, up_src0, up_tgt0, bufA, E0);
    launch_gemm(stream, bufA, nullptr, nullptr, nullptr, wt[0], P(LB[0]), bufB, SLOT(0), N0, 128);
    FIN(0, LG[0], LBE[0], N0);
    launch_gemm(stream, bufB, SLOT(0), nullptr, nullptr, wt[1], P(LB[1]), bufA, SLOT(1), N0, 128);
    FIN(1, LG[1], LBE[1], N0);
    launch_gemm(stream, bufA, SLOT(1), nullptr, nullptr, wt[2], P(LB[2]), bufB, SLOT(2), N0, 128);
    FIN(2, LG[2], LBE[2], N0);
    affine_relu_kernel<<<dim3((unsigned)(((long)N0 * 32 + 255) / 256)), 256, 0, stream>>>(
        bufB, SLOT(2), y0, (long)N0 * 32);

    // ---- dim 1: up1 = x1 + seg(x1[up_src1]) ; fc1 = x1 + seg(x0[face_src1]) ----
    launch_copy(stream, x1, bufA, (long)N1 * D);
    launch_scatter(stream, x1, up_src1, up_tgt1, bufA, E1);
    launch_copy(stream, x1, bufB, (long)N1 * D);
    launch_scatter(stream, x0, face_src1, face_tgt1, bufB, F1);
    // u1 branch
    launch_gemm(stream, bufA, nullptr, nullptr, nullptr, wt[3], P(LB[3]), bufC, SLOT(3), N1, 128);
    FIN(3, LG[3], LBE[3], N1);
    launch_gemm(stream, bufC, SLOT(3), nullptr, nullptr, wt[4], P(LB[4]), bufA, SLOT(4), N1, 128);
    FIN(4, LG[4], LBE[4], N1);                            // Hu2 in bufA, affine slot 4
    // f1 branch
    launch_gemm(stream, bufB, nullptr, nullptr, nullptr, wt[5], P(LB[5]), bufC, SLOT(5), N1, 128);
    FIN(5, LG[5], LBE[5], N1);
    launch_gemm(stream, bufC, SLOT(5), nullptr, nullptr, wt[6], P(LB[6]), bufB, SLOT(6), N1, 128);
    FIN(6, LG[6], LBE[6], N1);                            // Hf2 in bufB, affine slot 6
    // combine c1 on concat (K=256)
    launch_gemm(stream, bufA, SLOT(4), bufB, SLOT(6), wt[7], P(LB[7]), bufC, SLOT(7), N1, 256);
    FIN(7, LG[7], LBE[7], N1);
    affine_relu_kernel<<<dim3((unsigned)(((long)N1 * 32 + 255) / 256)), 256, 0, stream>>>(
        bufC, SLOT(7), y1, (long)N1 * 32);

    // ---- dim 2: fc2 = x2 + seg(x1[face_src2]) ----
    launch_copy(stream, x2, bufA, (long)N2 * D);
    launch_scatter(stream, x1, face_src2, face_tgt2, bufA, F2);
    launch_gemm(stream, bufA, nullptr, nullptr, nullptr, wt[8], P(LB[8]), bufB, SLOT(8), N2, 128);
    FIN(8, LG[8], LBE[8], N2);
    launch_gemm(stream, bufB, SLOT(8), nullptr, nullptr, wt[9], P(LB[9]), bufA, SLOT(9), N2, 128);
    FIN(9, LG[9], LBE[9], N2);
    launch_gemm(stream, bufA, SLOT(9), nullptr, nullptr, wt[10], P(LB[10]), bufB, SLOT(10), N2, 128);
    FIN(10, LG[10], LBE[10], N2);
    affine_relu_kernel<<<dim3((unsigned)(((long)N2 * 32 + 255) / 256)), 256, 0, stream>>>(
        bufB, SLOT(10), y2, (long)N2 * 32);
}
